// GradientLayer_29180007809717
// MI455X (gfx1250) — matrix operand order repaired and checked by reading, not compiled, not run on hardware
//
#include <hip/hip_runtime.h>

// CDNA5 / gfx1250: wave32, WMMA f32_16x16x32_f16 matrix path.
typedef _Float16 h8  __attribute__((ext_vector_type(8)));
typedef _Float16 h16 __attribute__((ext_vector_type(16)));
typedef float    f8  __attribute__((ext_vector_type(8)));

#define HD      128
#define TILE_M  16
#define WAVES   4

// ---- LDS layout (bytes) ----
// weights:  W2h(32K) W3h(32K) W2T(32K) W3T(32K) W1f(1536) W4f(512) b1(512) b2(512) b3(512)
#define OFF_WEIGHTS_END 134656
// per-wave: 9 x (16x128 f16 = 4096B) + xs(192) + cSt(64) + gSt(192) + hvSt(64)
#define WAVE_STRIDE     37376
#define SMEM_BYTES      (OFF_WEIGHTS_END + WAVES * WAVE_STRIDE)   // 284160 < 320KB WGP LDS

// Branch-free tanh: tanh(x) = sign(x) * (1 - e) * rcp(1 + e), e = exp(-2|x|).
// One v_exp_f32 + one v_rcp_f32 + FMAs; no EXEC divergence, no IEEE-divide
// expansion, so the epilogue co-executes with the (TRANS-tracked) f16 WMMA pipe.
__device__ __forceinline__ float fast_tanh(float x) {
  const float a = __builtin_fabsf(x);
  const float e = __expf(-2.0f * a);                 // v_exp_f32; ->0 for large a
  const float r = (1.0f - e) * __builtin_amdgcn_rcpf(1.0f + e);  // v_rcp_f32
  return __builtin_copysignf(r, x);
}

// A fragment: 16x32 f16 tile, row-major in LDS (stride HD).
// ISA layout: lane L -> M = L&15, K-base = (L>=16)?8:0 ; halves 0..7 = K=kb..kb+7,
// halves 8..15 = K=16+kb..23+kb  -> two contiguous ds_load_b128 per lane.
__device__ __forceinline__ h16 load_a_frag(const _Float16* A, int lane, int kc) {
  const int m  = lane & 15;
  const int kb = (lane >> 4) << 3;
  const _Float16* p = A + m * HD + kc * 32 + kb;
  h8 lo = *(const h8*)(p);
  h8 hi = *(const h8*)(p + 16);
  return __builtin_shufflevector(lo, hi, 0,1,2,3,4,5,6,7,8,9,10,11,12,13,14,15);
}

// B fragment: 32x16 f16 tile from row-major [K][N] weights (stride HD).
// Lane L -> N = L&15, K-base = (L>=16)?8:0 ; halves 0..7 = K=kb..kb+7, halves 8..15 = K=16+kb..23+kb.
__device__ __forceinline__ h16 load_b_frag(const _Float16* B, int lane, int kc, int n0) {
  const int kb = (lane >> 4) << 3;
  const _Float16* p = B + (kc * 32 + kb) * HD + n0 + (lane & 15);
  h16 v;
#pragma unroll
  for (int s = 0; s < 8; ++s) { v[s] = p[s * HD]; v[s + 8] = p[(s + 16) * HD]; }
  return v;
}

// 16x128 = (16x128) @ (128x128) GEMM: 8 n-tiles x 4 k-chunks of v_wmma_f32_16x16x32_f16,
// fused elementwise epilogue per output element.
template <class Epi>
__device__ __forceinline__ void gemm_16xHD(const _Float16* A, const _Float16* B,
                                           int lane, Epi epi) {
  const h16 a0 = load_a_frag(A, lane, 0);
  const h16 a1 = load_a_frag(A, lane, 1);
  const h16 a2 = load_a_frag(A, lane, 2);
  const h16 a3 = load_a_frag(A, lane, 3);
  for (int n0 = 0; n0 < HD; n0 += 16) {
    f8 acc = {};
    acc = __builtin_amdgcn_wmma_f32_16x16x32_f16(false, a0, false, load_b_frag(B, lane, 0, n0), (short)0, acc, false, false);
    acc = __builtin_amdgcn_wmma_f32_16x16x32_f16(false, a1, false, load_b_frag(B, lane, 1, n0), (short)0, acc, false, false);
    acc = __builtin_amdgcn_wmma_f32_16x16x32_f16(false, a2, false, load_b_frag(B, lane, 2, n0), (short)0, acc, false, false);
    acc = __builtin_amdgcn_wmma_f32_16x16x32_f16(false, a3, false, load_b_frag(B, lane, 3, n0), (short)0, acc, false, false);
    const int Mb = (lane >> 4) << 3;       // C layout: lanes 16-31 hold M=8..15
    const int Nc = n0 + (lane & 15);
#pragma unroll
    for (int r = 0; r < 8; ++r) epi(Mb + r, Nc, acc[r]);
  }
}

__global__ __launch_bounds__(WAVES * 32)
void pinn_wmma_kernel(const float* __restrict__ x,
                      const float* __restrict__ W1, const float* __restrict__ b1,
                      const float* __restrict__ W2, const float* __restrict__ b2,
                      const float* __restrict__ W3, const float* __restrict__ b3,
                      const float* __restrict__ W4, const float* __restrict__ b4,
                      float* __restrict__ out, int N) {
  extern __shared__ char smem[];
  _Float16* W2h = (_Float16*)smem;           // [128][128]  W2[k][n]
  _Float16* W3h = W2h + HD * HD;             // W3[k][n]
  _Float16* W2T = W3h + HD * HD;             // W2T[k][n] = W2[n][k]
  _Float16* W3T = W2T + HD * HD;             // W3T[k][n] = W3[n][k]
  float* W1f = (float*)(W3T + HD * HD);      // [3][128]
  float* W4f = W1f + 3 * HD;                 // [128]
  float* b1f = W4f + HD;
  float* b2f = b1f + HD;
  float* b3f = b2f + HD;

  const int tid  = threadIdx.x;
  const int lane = tid & 31;
  const int wv   = tid >> 5;

  char* wb = smem + OFF_WEIGHTS_END + wv * WAVE_STRIDE;
  _Float16* H1 = (_Float16*)wb;      // activations (f16, 16x128 each)
  _Float16* H2 = H1 + 2048;
  _Float16* H3 = H2 + 2048;
  _Float16* G2 = H3 + 2048;          // pre-D backprop values (W3 u3)
  _Float16* G1 = G2 + 2048;          // (W2 u2)
  _Float16* U3 = G1 + 2048;          // u3; reused as th1 in tangent passes
  _Float16* U2 = U3 + 2048;          // u2; reused as th2
  _Float16* D3 = U2 + 2048;          // delta-u3
  _Float16* D2 = D3 + 2048;          // delta-u2
  float* xs   = (float*)(D2 + 2048); // [16][3] input tile
  float* cSt  = xs + 48;             // [16] value accum
  float* gSt  = cSt + 16;            // [16][3] gradient accum
  float* hvSt = gSt + 48;            // [16] HVP accum

  // ---- stage weights (f16 + explicit transposes) ----
  for (int i = tid; i < HD * HD; i += blockDim.x) {
    W2h[i] = (_Float16)W2[i];
    W3h[i] = (_Float16)W3[i];
    const int r = i >> 7, c = i & 127;
    W2T[i] = (_Float16)W2[c * HD + r];
    W3T[i] = (_Float16)W3[c * HD + r];
  }
  for (int i = tid; i < 3 * HD; i += blockDim.x) W1f[i] = W1[i];
  for (int i = tid; i < HD; i += blockDim.x) {
    W4f[i] = W4[i]; b1f[i] = b1[i]; b2f[i] = b2[i]; b3f[i] = b3[i];
  }

  const int sbase = blockIdx.x * (WAVES * TILE_M) + wv * TILE_M;
  if (lane < 16) {
    const float* xp = x + (size_t)(sbase + lane) * 3;
    xs[lane * 3 + 0] = xp[0]; xs[lane * 3 + 1] = xp[1]; xs[lane * 3 + 2] = xp[2];
    cSt[lane] = 0.f;
    gSt[lane * 3 + 0] = 0.f; gSt[lane * 3 + 1] = 0.f; gSt[lane * 3 + 2] = 0.f;
  }
  __syncthreads();

  // ---- layer 1 (K=3, scalar) : h1 = tanh(x@W1 + b1) ----
  for (int e = 0; e < 64; ++e) {
    const int f = e * 32 + lane; const int m = f >> 7, n = f & 127;
    float z = xs[m*3+0]*W1f[n] + xs[m*3+1]*W1f[HD+n] + xs[m*3+2]*W1f[2*HD+n] + b1f[n];
    H1[f] = (_Float16)fast_tanh(z);
  }
  __syncthreads();

  // ---- forward: h2, h3, c ----
  gemm_16xHD(H1, W2h, lane, [&](int M, int Nc, float acc) {
    H2[M * HD + Nc] = (_Float16)fast_tanh(acc + b2f[Nc]);
  });
  __syncthreads();
  gemm_16xHD(H2, W3h, lane, [&](int M, int Nc, float acc) {
    const float v = fast_tanh(acc + b3f[Nc]);
    H3[M * HD + Nc] = (_Float16)v;
    atomicAdd(&cSt[M], v * W4f[Nc]);
  });
  __syncthreads();

  // ---- reverse: u3 = (1-h3^2) w4 ; u2 = D2 (u3@W3^T) ; u1 = D1 (u2@W2^T) ; g = u1@W1^T ----
  for (int e = 0; e < 64; ++e) {
    const int f = e * 32 + lane; const int n = f & 127;
    const float h3 = (float)H3[f];
    U3[f] = (_Float16)((1.f - h3 * h3) * W4f[n]);
  }
  __syncthreads();
  gemm_16xHD(U3, W3T, lane, [&](int M, int Nc, float acc) {
    G2[M * HD + Nc] = (_Float16)acc;
    const float h2 = (float)H2[M * HD + Nc];
    U2[M * HD + Nc] = (_Float16)((1.f - h2 * h2) * acc);
  });
  __syncthreads();
  gemm_16xHD(U2, W2T, lane, [&](int M, int Nc, float acc) {
    G1[M * HD + Nc] = (_Float16)acc;
    const float h1 = (float)H1[M * HD + Nc];
    const float u1 = (1.f - h1 * h1) * acc;
    atomicAdd(&gSt[M * 3 + 0], u1 * W1f[Nc]);
    atomicAdd(&gSt[M * 3 + 1], u1 * W1f[HD + Nc]);
    atomicAdd(&gSt[M * 3 + 2], u1 * W1f[2 * HD + Nc]);
  });
  __syncthreads();

  if (lane < 16) {
    const int s = sbase + lane;
    out[s]                 = cSt[lane] + b4[0];   // c
    out[(size_t)N + s]     = gSt[lane * 3 + 0];   // dc_dt
    out[(size_t)2 * N + s] = gSt[lane * 3 + 1];   // dc_dx
    out[(size_t)3 * N + s] = gSt[lane * 3 + 2];   // dc_dy
  }

  // ---- HVP (forward-over-reverse) for e_x (j=1) and e_y (j=2) ----
  for (int dir = 0; dir < 2; ++dir) {
    const int j = dir + 1;
    const float* W1j = W1f + j * HD;
    if (lane < 16) hvSt[lane] = 0.f;
    // th1 = D1 * W1[j,:]
    for (int e = 0; e < 64; ++e) {
      const int f = e * 32 + lane; const int n = f & 127;
      const float h1 = (float)H1[f];
      U3[f] = (_Float16)((1.f - h1 * h1) * W1j[n]);   // th1
    }
    __syncthreads();
    gemm_16xHD(U3, W2h, lane, [&](int M, int Nc, float acc) {   // t2 = th1@W2
      const float h2 = (float)H2[M * HD + Nc];
      U2[M * HD + Nc] = (_Float16)((1.f - h2 * h2) * acc);      // th2
    });
    __syncthreads();
    gemm_16xHD(U2, W3h, lane, [&](int M, int Nc, float acc) {   // t3 = th2@W3
      const float h3 = (float)H3[M * HD + Nc];
      const float th3 = (1.f - h3 * h3) * acc;
      D3[M * HD + Nc] = (_Float16)(-2.f * h3 * th3 * W4f[Nc]);  // du3
    });
    __syncthreads();
    gemm_16xHD(D3, W3T, lane, [&](int M, int Nc, float acc) {   // du3@W3^T
      const float h2  = (float)H2[M * HD + Nc];
      const float th2 = (float)U2[M * HD + Nc];
      const float g2  = (float)G2[M * HD + Nc];
      D2[M * HD + Nc] = (_Float16)(-2.f * h2 * th2 * g2 + (1.f - h2 * h2) * acc);
    });
    __syncthreads();
    gemm_16xHD(D2, W2T, lane, [&](int M, int Nc, float acc) {   // du2@W2^T
      const float h1  = (float)H1[M * HD + Nc];
      const float th1 = (float)U3[M * HD + Nc];
      const float g1  = (float)G1[M * HD + Nc];
      const float du1 = -2.f * h1 * th1 * g1 + (1.f - h1 * h1) * acc;
      atomicAdd(&hvSt[M], du1 * W1j[Nc]);                       // (Hv)[j]
    });
    __syncthreads();
    if (lane < 16) out[(size_t)(4 + dir) * N + sbase + lane] = hvSt[lane];
    __syncthreads();
  }
}

extern "C" void kernel_launch(void* const* d_in, const int* in_sizes, int n_in,
                              void* d_out, int out_size, void* d_ws, size_t ws_size,
                              hipStream_t stream) {
  (void)n_in; (void)out_size; (void)d_ws; (void)ws_size;
  const float* x  = (const float*)d_in[0];
  const float* W1 = (const float*)d_in[1];
  const float* b1 = (const float*)d_in[2];
  const float* W2 = (const float*)d_in[3];
  const float* b2 = (const float*)d_in[4];
  const float* W3 = (const float*)d_in[5];
  const float* b3 = (const float*)d_in[6];
  const float* W4 = (const float*)d_in[7];
  const float* b4 = (const float*)d_in[8];
  float* out = (float*)d_out;
  const int N = in_sizes[0] / 3;                      // 262144
  const int blocks = N / (WAVES * TILE_M);            // 4096
  pinn_wmma_kernel<<<blocks, WAVES * 32, SMEM_BYTES, stream>>>(
      x, W1, b1, W2, b2, W3, b3, W4, b4, out, N);
}
